// AdaLNBlock_87033217286787
// MI455X (gfx1250) — compile-verified, matrix operand order checked by reading, not run on hardware
//
#include <hip/hip_runtime.h>
#include <hip/hip_bf16.h>
#include <math.h>

typedef __attribute__((ext_vector_type(16))) __bf16 v16bf;
typedef __attribute__((ext_vector_type(8)))  float  v8f;

#define D     1024
#define NSEQ  2048
#define BATCH 4
#define DFF   4096
#define HD    64
#define HKV   4
#define KVW   (HKV*HD)        // 256
#define MROWS (BATCH*NSEQ)    // 8192
#define SIXD  (6*D)

union FragU { v16bf v; uint4 q[2]; };

// Both A (16x32, lane=m) and B (32x16, lane=n) bf16 fragments load K-major:
// per-lane: row = lane&15, K chunk0 at koff, chunk1 at koff+16, koff = 8*(lane>=16)
__device__ __forceinline__ v16bf load_frag_kmajor(const __bf16* base, int ld) {
  int lane = threadIdx.x & 31;
  const __bf16* p = base + (size_t)(lane & 15) * ld + ((lane >> 4) << 3);
  FragU f;
  f.q[0] = *(const uint4*)(p);
  f.q[1] = *(const uint4*)(p + 16);
  return f.v;
}

__device__ __forceinline__ v8f wmma_bf16(v16bf a, v16bf b, v8f c) {
  return __builtin_amdgcn_wmma_f32_16x16x32_bf16(false, a, false, b, (short)0, c,
                                                 false, false);
}

// ---- CDNA5 async global->LDS copy (ASYNCcnt-tracked, bypasses VGPRs) ----
__device__ __forceinline__ void async_load_b128(unsigned lds_off, const void* gptr) {
  asm volatile("global_load_async_to_lds_b128 %0, %1, off"
               :: "v"(lds_off), "v"((unsigned long long)(size_t)gptr)
               : "memory");
}
__device__ __forceinline__ void wait_async_8() {
  asm volatile("s_wait_asynccnt 0x8" ::: "memory");
}
__device__ __forceinline__ void wait_async_0() {
  asm volatile("s_wait_asynccnt 0x0" ::: "memory");
}

__device__ __forceinline__ float half_reduce_max(float v) {
  #pragma unroll
  for (int o = 1; o < 16; o <<= 1) v = fmaxf(v, __shfl_xor(v, o, 32));
  return v;
}
__device__ __forceinline__ float half_reduce_sum(float v) {
  #pragma unroll
  for (int o = 1; o < 16; o <<= 1) v += __shfl_xor(v, o, 32);
  return v;
}

__device__ __forceinline__ float blockReduceSum(float val) {
  __shared__ float red[8];
  __syncthreads();
  #pragma unroll
  for (int o = 16; o > 0; o >>= 1) val += __shfl_xor(val, o, 32);
  int wid = threadIdx.x >> 5, lane = threadIdx.x & 31;
  if (lane == 0) red[wid] = val;
  __syncthreads();
  float tot = 0.f;
  #pragma unroll
  for (int i = 0; i < 8; i++) tot += red[i];
  return tot;
}

// ---------------- small kernels ----------------

__global__ void f32_to_bf16(const float* __restrict__ in, __bf16* __restrict__ out, int n) {
  for (int i = blockIdx.x * 256 + threadIdx.x; i < n; i += gridDim.x * 256)
    out[i] = (__bf16)in[i];
}

// cond = silu(c) @ wc.T + bc  : (B, 6D), tiny M so VALU is fine
__global__ void cond_kernel(const float* __restrict__ c, const float* __restrict__ wc,
                            const float* __restrict__ bc, float* __restrict__ cond) {
  int j = blockIdx.x * 256 + threadIdx.x;          // 0 .. B*6D
  int b = j / SIXD, col = j % SIXD;
  const float* cr = c + (size_t)b * D;
  const float* wr = wc + (size_t)col * D;
  float acc = 0.f;
  for (int k = 0; k < D; k++) {
    float cv = cr[k];
    acc += (cv / (1.f + __expf(-cv))) * wr[k];
  }
  cond[(size_t)b * SIXD + col] = acc + bc[col];
}

// h = LN(x) * (1+sc) + sh  -> bf16, one block per (b,n) row, 256 thr * 4 elems
__global__ __launch_bounds__(256) void ln_mod_kernel(const float* __restrict__ x,
                                                     const float* __restrict__ cond,
                                                     int sh_off, int sc_off,
                                                     __bf16* __restrict__ h) {
  int row = blockIdx.x;
  int b = row / NSEQ;
  const float* xr = x + (size_t)row * D;
  int tid = threadIdx.x;
  float v[4], s = 0.f;
  #pragma unroll
  for (int i = 0; i < 4; i++) { v[i] = xr[tid + i * 256]; s += v[i]; }
  float mean = blockReduceSum(s) * (1.f / (float)D);
  float s2 = 0.f;
  #pragma unroll
  for (int i = 0; i < 4; i++) { float d = v[i] - mean; s2 += d * d; }
  float rstd = rsqrtf(blockReduceSum(s2) * (1.f / (float)D) + 1e-5f);
  const float* cb = cond + (size_t)b * SIXD;
  #pragma unroll
  for (int i = 0; i < 4; i++) {
    int col = tid + i * 256;
    float sc = cb[sc_off + col], sh = cb[sh_off + col];
    h[(size_t)row * D + col] = (__bf16)(((v[i] - mean) * rstd) * (1.f + sc) + sh);
  }
}

// ---------------- WMMA GEMM: out = epi(A @ Bw.T) ----------------
// A: (M,K) bf16 row-major, Bw: (Nn,K) bf16 row-major. 128x128 block tile,
// BK=64, double-buffered LDS filled by async global->LDS b128 copies.
// 8 waves in 4x2; each wave 32x64 (2x4 WMMA 16x16x32 accumulators),
// 16 WMMAs per K-iteration per wave.
// EPI 0: store bf16 ; 1: f32 resid + gate*acc ; 2: bias+gelu->bf16 ;
// EPI 3: f32 resid + gate*(acc+bias)
template <int EPI>
__global__ __launch_bounds__(256) void gemm_wmma(
    const __bf16* __restrict__ A, const __bf16* __restrict__ Bw,
    const float* __restrict__ bias, const float* __restrict__ resid,
    const float* __restrict__ gate, int gate_ld, void* __restrict__ outp,
    int M, int Nn, int K, int n_per_b) {
  __shared__ __bf16 As[2][128][64];   // 32 KB
  __shared__ __bf16 Bs[2][128][64];   // 32 KB
  const int tid = threadIdx.x;
  const int lane = tid & 31;
  const int wid = tid >> 5;
  const int wm = wid & 3, wn = wid >> 2;
  const int m0 = blockIdx.y * 128, n0 = blockIdx.x * 128;

  v8f acc[2][4];
  #pragma unroll
  for (int i = 0; i < 2; i++)
    #pragma unroll
    for (int j = 0; j < 4; j++) { v8f z = {}; acc[i][j] = z; }

  // issue one 128x64 A tile + 128x64 B tile as 8 async b128 copies per thread
  auto issue = [&](int k0, int buf) {
    #pragma unroll
    for (int l = 0; l < 4; l++) {
      int cidx = tid + l * 256;                 // 0..1023 chunks of 8 bf16
      int r = cidx >> 3, col = (cidx & 7) << 3;
      async_load_b128((unsigned)(size_t)&As[buf][r][col],
                      A + (size_t)(m0 + r) * K + k0 + col);
    }
    #pragma unroll
    for (int l = 0; l < 4; l++) {
      int cidx = tid + l * 256;
      int r = cidx >> 3, col = (cidx & 7) << 3;
      async_load_b128((unsigned)(size_t)&Bs[buf][r][col],
                      Bw + (size_t)(n0 + r) * K + k0 + col);
    }
  };

  const int kt = K >> 6;
  issue(0, 0);
  for (int k = 0; k < kt; ++k) {
    const int buf = k & 1;
    if (k + 1 < kt) {
      issue((k + 1) << 6, buf ^ 1);
      wait_async_8();                 // tile k's 8 copies (in-order) complete
    } else {
      wait_async_0();
    }
    __syncthreads();                  // all waves' copies of tile k visible
    #pragma unroll
    for (int ks = 0; ks < 2; ks++) {  // two K=32 steps inside the 64-wide tile
      v16bf a0 = load_frag_kmajor(&As[buf][wm * 32][ks * 32], 64);
      v16bf a1 = load_frag_kmajor(&As[buf][wm * 32 + 16][ks * 32], 64);
      #pragma unroll
      for (int j = 0; j < 4; j++) {
        v16bf bfr = load_frag_kmajor(&Bs[buf][wn * 64 + j * 16][ks * 32], 64);
        acc[0][j] = wmma_bf16(a0, bfr, acc[0][j]);
        acc[1][j] = wmma_bf16(a1, bfr, acc[1][j]);
      }
    }
    __syncthreads();                  // reads done before buf is re-filled
  }

  const int hi = (lane >> 4) ? 8 : 0;
  const int cl = lane & 15;
  #pragma unroll
  for (int i = 0; i < 2; i++)
    #pragma unroll
    for (int j = 0; j < 4; j++)
      #pragma unroll
      for (int r = 0; r < 8; r++) {
        int row = m0 + wm * 32 + i * 16 + r + hi;
        int col = n0 + wn * 64 + j * 16 + cl;
        float v = acc[i][j][r];
        size_t idx = (size_t)row * Nn + col;
        if constexpr (EPI == 0) {
          ((__bf16*)outp)[idx] = (__bf16)v;
        } else if constexpr (EPI == 1) {
          int bb = row / n_per_b;
          ((float*)outp)[idx] = resid[idx] + gate[(size_t)bb * gate_ld + col] * v;
        } else if constexpr (EPI == 2) {
          v += bias[col];
          v = 0.5f * v * (1.f + erff(v * 0.70710678118654752f));
          ((__bf16*)outp)[idx] = (__bf16)v;
        } else {
          v += bias[col];
          int bb = row / n_per_b;
          ((float*)outp)[idx] = resid[idx] + gate[(size_t)bb * gate_ld + col] * v;
        }
      }
}

// ---------------- flash attention (GQA), bf16 WMMA ----------------
// grid: (NSEQ/128, B*H). Each block: one (b, kv-head g, rep r), 128 q rows.
// Each wave: 16 q rows, streams 32-key tiles; scores via WMMA from global K
// (row-major K == B-fragment K-major); V transposed through LDS for PV.
__global__ __launch_bounds__(256) void attn_kernel(const __bf16* __restrict__ qb,
                                                   const __bf16* __restrict__ kb,
                                                   const __bf16* __restrict__ vb,
                                                   __bf16* __restrict__ ob) {
  __shared__ __bf16 Vt[64][32];       // V^T tile: [dim][key]
  __shared__ __bf16 Pl[8][16][32];    // per-wave P tile (q x key)
  const int tid = threadIdx.x;
  const int lane = tid & 31;
  const int wid = tid >> 5;
  const int bgr = blockIdx.y;
  const int b = bgr >> 4;
  const int hrem = bgr & 15;          // g*REP + r
  const int g = hrem >> 2;
  const int hcol = hrem * HD;         // column in D for q/out
  const int kvcol = g * HD;           // column in KVW for k/v
  const int q0 = blockIdx.x * 128 + wid * 16;
  const size_t qrowbase = (size_t)b * NSEQ + q0;

  v16bf qf0 = load_frag_kmajor(qb + qrowbase * D + hcol, D);
  v16bf qf1 = load_frag_kmajor(qb + qrowbase * D + hcol + 32, D);

  float mrow[8], lrow[8];
  v8f o[4];
  #pragma unroll
  for (int r = 0; r < 8; r++) { mrow[r] = -1e30f; lrow[r] = 0.f; }
  #pragma unroll
  for (int t = 0; t < 4; t++) { v8f z = {}; o[t] = z; }

  const int hi = (lane >> 4) ? 8 : 0;
  const int cl = lane & 15;

  for (int key0 = 0; key0 < NSEQ; key0 += 32) {
    __syncthreads();
    {  // cooperative transposed V tile load: 32 keys x 64 dims
      int key = tid >> 3;
      int dg = (tid & 7) << 3;
      const __bf16* vp = vb + (size_t)(b * NSEQ + key0 + key) * KVW + kvcol + dg;
      #pragma unroll
      for (int i = 0; i < 8; i++) Vt[dg + i][key] = vp[i];
    }
    // S = Q (16x64) @ K^T (64x32) from global, two 16-key C tiles
    const __bf16* kbase = kb + (size_t)(b * NSEQ + key0) * KVW + kvcol;
    v8f s0 = {}, s1 = {};
    s0 = wmma_bf16(qf0, load_frag_kmajor(kbase, KVW), s0);
    s0 = wmma_bf16(qf1, load_frag_kmajor(kbase + 32, KVW), s0);
    s1 = wmma_bf16(qf0, load_frag_kmajor(kbase + 16 * KVW, KVW), s1);
    s1 = wmma_bf16(qf1, load_frag_kmajor(kbase + 16 * KVW + 32, KVW), s1);

    #pragma unroll
    for (int r = 0; r < 8; r++) {       // online softmax per q row
      float t0 = s0[r] * 0.125f;        // 1/sqrt(64)
      float t1 = s1[r] * 0.125f;
      float mnew = fmaxf(mrow[r], half_reduce_max(fmaxf(t0, t1)));
      float p0 = __expf(t0 - mnew);
      float p1 = __expf(t1 - mnew);
      float rs = half_reduce_sum(p0 + p1);
      float alpha = __expf(mrow[r] - mnew);
      lrow[r] = lrow[r] * alpha + rs;
      mrow[r] = mnew;
      #pragma unroll
      for (int t = 0; t < 4; t++) o[t][r] *= alpha;
      Pl[wid][r + hi][cl] = (__bf16)p0;
      Pl[wid][r + hi][16 + cl] = (__bf16)p1;
    }
    __syncthreads();
    v16bf pf = load_frag_kmajor(&Pl[wid][0][0], 32);   // P as A frag (16x32)
    #pragma unroll
    for (int t = 0; t < 4; t++) {
      v16bf vf = load_frag_kmajor(&Vt[t * 16][0], 32); // V^T as B frag
      o[t] = wmma_bf16(pf, vf, o[t]);
    }
  }

  #pragma unroll
  for (int t = 0; t < 4; t++)
    #pragma unroll
    for (int r = 0; r < 8; r++) {
      size_t row = qrowbase + r + hi;
      ob[row * D + hcol + t * 16 + cl] = (__bf16)(o[t][r] / lrow[r]);
    }
}

// ---------------- host ----------------

extern "C" void kernel_launch(void* const* d_in, const int* in_sizes, int n_in,
                              void* d_out, int out_size, void* d_ws, size_t ws_size,
                              hipStream_t stream) {
  const float* x  = (const float*)d_in[0];
  const float* c  = (const float*)d_in[1];
  const float* wq = (const float*)d_in[2];
  const float* wk = (const float*)d_in[3];
  const float* wv = (const float*)d_in[4];
  const float* wo = (const float*)d_in[5];
  const float* w1 = (const float*)d_in[6];
  const float* b1 = (const float*)d_in[7];
  const float* w2 = (const float*)d_in[8];
  const float* b2 = (const float*)d_in[9];
  const float* wc = (const float*)d_in[10];
  const float* bc = (const float*)d_in[11];
  float* out = (float*)d_out;

  char* ws = (char*)d_ws;
  auto take = [&](size_t bytes) {
    char* p = ws;
    ws += (bytes + 255) & ~(size_t)255;
    return p;
  };
  __bf16* wq_b  = (__bf16*)take((size_t)D * D * 2);
  __bf16* wk_b  = (__bf16*)take((size_t)KVW * D * 2);
  __bf16* wv_b  = (__bf16*)take((size_t)KVW * D * 2);
  __bf16* wo_b  = (__bf16*)take((size_t)D * D * 2);
  __bf16* w1_b  = (__bf16*)take((size_t)DFF * D * 2);
  __bf16* w2_b  = (__bf16*)take((size_t)D * DFF * 2);
  float*  cond  = (float*)take((size_t)BATCH * SIXD * 4);
  __bf16* h     = (__bf16*)take((size_t)MROWS * D * 2);
  __bf16* qbuf  = (__bf16*)take((size_t)MROWS * D * 2);
  __bf16* kbuf  = (__bf16*)take((size_t)MROWS * KVW * 2);
  __bf16* vbuf  = (__bf16*)take((size_t)MROWS * KVW * 2);
  __bf16* attnb = (__bf16*)take((size_t)MROWS * D * 2);
  float*  xmid  = (float*)take((size_t)MROWS * D * 4);
  __bf16* mlp   = (__bf16*)take((size_t)MROWS * DFF * 2);

  // 1) weights -> bf16
  f32_to_bf16<<<256, 256, 0, stream>>>(wq, wq_b, D * D);
  f32_to_bf16<<<256, 256, 0, stream>>>(wk, wk_b, KVW * D);
  f32_to_bf16<<<256, 256, 0, stream>>>(wv, wv_b, KVW * D);
  f32_to_bf16<<<256, 256, 0, stream>>>(wo, wo_b, D * D);
  f32_to_bf16<<<512, 256, 0, stream>>>(w1, w1_b, DFF * D);
  f32_to_bf16<<<512, 256, 0, stream>>>(w2, w2_b, D * DFF);

  // 2) AdaLN conditioning
  cond_kernel<<<(BATCH * SIXD) / 256, 256, 0, stream>>>(c, wc, bc, cond);

  // 3) h = LN(x)*(1+sc1)+sh1
  ln_mod_kernel<<<MROWS, 256, 0, stream>>>(x, cond, 0 * D, 1 * D, h);

  // 4) q/k/v projections
  gemm_wmma<0><<<dim3(D / 128, MROWS / 128), 256, 0, stream>>>(
      h, wq_b, nullptr, nullptr, nullptr, 0, qbuf, MROWS, D, D, NSEQ);
  gemm_wmma<0><<<dim3(KVW / 128, MROWS / 128), 256, 0, stream>>>(
      h, wk_b, nullptr, nullptr, nullptr, 0, kbuf, MROWS, KVW, D, NSEQ);
  gemm_wmma<0><<<dim3(KVW / 128, MROWS / 128), 256, 0, stream>>>(
      h, wv_b, nullptr, nullptr, nullptr, 0, vbuf, MROWS, KVW, D, NSEQ);

  // 5) flash attention
  attn_kernel<<<dim3(NSEQ / 128, BATCH * 16), 256, 0, stream>>>(qbuf, kbuf, vbuf, attnb);

  // 6) xmid = x + g1 * (attn @ wo.T)
  gemm_wmma<1><<<dim3(D / 128, MROWS / 128), 256, 0, stream>>>(
      attnb, wo_b, nullptr, x, cond + 2 * D, SIXD, xmid, MROWS, D, D, NSEQ);

  // 7) h2 = LN(xmid)*(1+sc2)+sh2
  ln_mod_kernel<<<MROWS, 256, 0, stream>>>(xmid, cond, 3 * D, 4 * D, h);

  // 8) mlp = gelu(h2 @ w1.T + b1)
  gemm_wmma<2><<<dim3(DFF / 128, MROWS / 128), 256, 0, stream>>>(
      h, w1_b, b1, nullptr, nullptr, 0, mlp, MROWS, DFF, D, NSEQ);

  // 9) out = xmid + g2 * (mlp @ w2.T + b2)
  gemm_wmma<3><<<dim3(D / 128, MROWS / 128), 256, 0, stream>>>(
      mlp, w2_b, b2, xmid, cond + 5 * D, SIXD, out, MROWS, D, DFF, NSEQ);
}
